// GCNClusterNet_16303695856283
// MI455X (gfx1250) — matrix operand order repaired and checked by reading, not compiled, not run on hardware
//
#include <hip/hip_runtime.h>
#include <hip/hip_bf16.h>

// ---------------------------------------------------------------------------
// GCNClusterNet on MI455X (gfx1250): bf16 WMMA GEMM pipeline + f32 k-means.
// N=10000, NFEAT=1024, NHID=512, NOUT=128, KCLUST=50, num_iter=10 (fixed).
// Roofline: adj (400 MB f32) streamed once per adj-GEMM (block spans full N);
// bf16 WMMA 16x16x32, f32 accumulation; 1 A-fragment feeds 4 WMMAs; k-loop is
// double-buffered with a straight-line paired body (no mid-loop branches) so
// accumulators stay pinned and next step's loads overlap current WMMAs.
// ---------------------------------------------------------------------------

typedef __attribute__((ext_vector_type(16))) __bf16 v16bf;
typedef __attribute__((ext_vector_type(8)))  float  v8f;

#define KCLUST 50
#define DOUT   128

// ---------------------------------------------------------------------------
// Transpose + convert f32 [K x N] -> bf16 [N x KP] (zero-padded K -> KP).
// ---------------------------------------------------------------------------
__global__ void k_transpose_bf16(const float* __restrict__ B, __bf16* __restrict__ Bt,
                                 int K, int N, int KP)
{
    long idx   = (long)blockIdx.x * blockDim.x + threadIdx.x;
    long total = (long)N * KP;
    if (idx >= total) return;
    int n = (int)(idx / KP);
    int k = (int)(idx % KP);
    Bt[idx] = (k < K) ? (__bf16)B[(size_t)k * N + n] : (__bf16)0.0f;
}

// One k-step's in-flight data: raw f32 A (converted at use) + 4 B fragments.
struct KStep {
    float4 q0, q1, q2, q3;
    v16bf  b[4];
};

__device__ __forceinline__ void issue_loads(const float* __restrict__ Arow,
                                            const __bf16* const (&Brow)[4],
                                            int k0, int half, KStep& s)
{
    const float4* p0 = reinterpret_cast<const float4*>(Arow + k0 + half * 16);
    const float4* p1 = reinterpret_cast<const float4*>(Arow + k0 + half * 16 + 8);
    s.q0 = p0[0]; s.q1 = p0[1]; s.q2 = p1[0]; s.q3 = p1[1];
    #pragma unroll
    for (int j = 0; j < 4; ++j)
        s.b[j] = *reinterpret_cast<const v16bf*>(Brow[j] + k0 + half * 16);
}

__device__ __forceinline__ void compute_step(const KStep& s, v8f (&acc)[4])
{
    // A fragment: contiguous k order (not AMD's table order). lanes 0-15: K = 0-15,
    // lanes 16-31: K = 16-31 (relative to k0), matching B's contiguous order.
    v16bf a;
    a[0]=(__bf16)s.q0.x;  a[1]=(__bf16)s.q0.y;  a[2]=(__bf16)s.q0.z;  a[3]=(__bf16)s.q0.w;
    a[4]=(__bf16)s.q1.x;  a[5]=(__bf16)s.q1.y;  a[6]=(__bf16)s.q1.z;  a[7]=(__bf16)s.q1.w;
    a[8]=(__bf16)s.q2.x;  a[9]=(__bf16)s.q2.y;  a[10]=(__bf16)s.q2.z; a[11]=(__bf16)s.q2.w;
    a[12]=(__bf16)s.q3.x; a[13]=(__bf16)s.q3.y; a[14]=(__bf16)s.q3.z; a[15]=(__bf16)s.q3.w;
    #pragma unroll
    for (int j = 0; j < 4; ++j)
        acc[j] = __builtin_amdgcn_wmma_f32_16x16x32_bf16(false, a, false, s.b[j],
                                                         (short)0, acc[j], false, false);
}

// ---------------------------------------------------------------------------
// WMMA GEMM: C[M x N] = A[M x K](f32 row-major, converted inline to bf16)
//                       @ B (staged Bt[N x KP] bf16 = B^T, K zero-padded)
//            + bias[N] (optional), optional ReLU.
// Each wave computes a 16(M) x 64(N) strip: one A fragment -> 4 WMMAs.
// Block = 8 waves arranged WM x WN (WM*WN == 8); block tile = 16*WM x 64*WN.
// Launch so 64*WN*gridDim.y == N: then A rows are read from DRAM exactly once.
// Output: mode 0 -> Cf f32 row-major (ldc); mode 1 -> Ct bf16 transposed
// (Ct[n*ldct + m]) for direct use as the next GEMM's Bt.
// ---------------------------------------------------------------------------
template<int WM, int WN>
__global__ __launch_bounds__(256)
void wmma_gemm(const float*  __restrict__ A,
               const __bf16* __restrict__ Bt,
               const float*  __restrict__ bias,
               float*        __restrict__ Cf,
               __bf16*       __restrict__ Ct,
               int Mm, int Nn, int Kk, int KPb, int ldc, int ldct,
               int relu, int mode)
{
    const int lane = threadIdx.x & 31;
    const int wave = threadIdx.x >> 5;          // 0..7
    const int half = lane >> 4;                 // 0 or 1
    const int colL = lane & 15;                 // 0..15
    const int mw = wave % WM;
    const int nw = wave / WM;
    const int m0 = (blockIdx.x * WM + mw) * 16;
    const int n0 = (blockIdx.y * WN + nw) * 64;
    if (m0 >= Mm || n0 >= Nn) return;           // wave-uniform: EXEC stays all-1s

    const float* Arow = A + (size_t)(m0 + colL) * (size_t)Kk;
    const __bf16* Brow[4];
    #pragma unroll
    for (int s = 0; s < 4; ++s)
        Brow[s] = Bt + (size_t)(n0 + s * 16 + colL) * (size_t)KPb;

    v8f acc[4] = {};
    const int Kfull = Kk & ~31;
    int k0 = 0;

    if (Kfull > 0) {
        KStep s0, s1;
        issue_loads(Arow, Brow, 0, half, s0);
        // paired, straight-line body: no branches between the two computes, so
        // the accumulators stay pinned (WMMA accumulates in place) and each
        // step's loads are issued a full step ahead of their use.
        while (k0 + 64 <= Kfull) {
            issue_loads(Arow, Brow, k0 + 32, half, s1);
            __builtin_prefetch((const void*)(Arow + k0 + 128), 0, 1); // stream A ahead
            compute_step(s0, acc);
            int kn = k0 + 64;
            if (kn + 32 > Kfull) kn = Kfull - 32;  // clamp (redundant reload on last iter)
            issue_loads(Arow, Brow, kn, half, s0);
            compute_step(s1, acc);
            k0 += 64;
        }
        if (k0 < Kfull) {  // odd number of 32-steps: one leftover, already loaded
            compute_step(s0, acc);
            k0 += 32;
        }
    }
    if (k0 < Kk) {  // K tail (K % 32 == 16 for the adj GEMMs): guarded A loads
        v16bf a;
        #pragma unroll
        for (int i = 0; i < 8; ++i) {
            int ka = k0 + half * 16 + i;
            int kb = k0 + half * 16 + 8 + i;
            a[i]     = (ka < Kk) ? (__bf16)Arow[ka] : (__bf16)0.0f;
            a[8 + i] = (kb < Kk) ? (__bf16)Arow[kb] : (__bf16)0.0f;
        }
        #pragma unroll
        for (int s = 0; s < 4; ++s) {
            v16bf b = *reinterpret_cast<const v16bf*>(Brow[s] + k0 + half * 16); // pad = 0
            acc[s] = __builtin_amdgcn_wmma_f32_16x16x32_bf16(false, a, false, b,
                                                             (short)0, acc[s], false, false);
        }
    }

    #pragma unroll
    for (int s = 0; s < 4; ++s) {
        const int col = n0 + s * 16 + colL;
        const float bv = bias ? bias[col] : 0.0f;
        #pragma unroll
        for (int g = 0; g < 8; ++g) {
            float v = acc[s][g] + bv;
            if (relu) v = fmaxf(v, 0.0f);
            const int row = m0 + half * 8 + g;  // C layout: VGPR g -> M = half*8+g
            if (mode == 0) Cf[(size_t)row * ldc + col] = v;
            else           Ct[(size_t)col * ldct + row] = (__bf16)v;
        }
    }
}

// ---------------------------------------------------------------------------
// data[i,:] = embeds[i,:] / ||embeds[i,:]||  (one 128-thread block per row)
// ---------------------------------------------------------------------------
__global__ void k_normalize(const float* __restrict__ embeds, float* __restrict__ data)
{
    const int i = blockIdx.x, t = threadIdx.x;
    __shared__ float s[DOUT];
    float v = embeds[(size_t)i * DOUT + t];
    s[t] = v * v;
    __syncthreads();
    for (int st = DOUT / 2; st > 0; st >>= 1) {
        if (t < st) s[t] += s[t + st];
        __syncthreads();
    }
    data[(size_t)i * DOUT + t] = v * rsqrtf(s[0]);
}

// ---------------------------------------------------------------------------
// Per-row: dist[i,k] = <data[i], mu[k]>; r[i,:] = softmax(30*dist[i,:]).
// One thread per row; mu (25 KB) stays resident in cache.
// ---------------------------------------------------------------------------
__global__ void k_assign(const float* __restrict__ data, const float* __restrict__ mu,
                         float* __restrict__ r, float* __restrict__ dist, int nrows)
{
    const int i = blockIdx.x * blockDim.x + threadIdx.x;
    if (i >= nrows) return;
    const float* row = data + (size_t)i * DOUT;
    float acc[KCLUST];
    #pragma unroll
    for (int k = 0; k < KCLUST; ++k) acc[k] = 0.0f;
    for (int d = 0; d < DOUT; ++d) {
        const float xv = row[d];
        #pragma unroll
        for (int k = 0; k < KCLUST; ++k) acc[k] = fmaf(xv, mu[k * DOUT + d], acc[k]);
    }
    if (dist) {
        #pragma unroll
        for (int k = 0; k < KCLUST; ++k) dist[(size_t)i * KCLUST + k] = acc[k];
    }
    float mx = -1e30f;
    #pragma unroll
    for (int k = 0; k < KCLUST; ++k) mx = fmaxf(mx, 30.0f * acc[k]);
    float s = 0.0f;
    #pragma unroll
    for (int k = 0; k < KCLUST; ++k) { acc[k] = __expf(30.0f * acc[k] - mx); s += acc[k]; }
    const float inv = 1.0f / s;
    #pragma unroll
    for (int k = 0; k < KCLUST; ++k) r[(size_t)i * KCLUST + k] = acc[k] * inv;
}

// cluster_r[k] = sum_i r[i,k]   (one block per cluster)
__global__ void k_cluster_r(const float* __restrict__ r, float* __restrict__ cr, int nrows)
{
    const int k = blockIdx.x;
    __shared__ float s[256];
    float acc = 0.0f;
    for (int i = threadIdx.x; i < nrows; i += 256) acc += r[(size_t)i * KCLUST + k];
    s[threadIdx.x] = acc;
    __syncthreads();
    for (int st = 128; st > 0; st >>= 1) {
        if ((int)threadIdx.x < st) s[threadIdx.x] += s[threadIdx.x + st];
        __syncthreads();
    }
    if (threadIdx.x == 0) cr[k] = s[0];
}

// cmean[k,d] = sum_i r[i,k] * data[i,d]   (block per cluster, thread per dim)
__global__ void k_cluster_mean(const float* __restrict__ r, const float* __restrict__ data,
                               float* __restrict__ cmean, int nrows)
{
    const int k = blockIdx.x, d = threadIdx.x;
    float acc = 0.0f;
    #pragma unroll 4
    for (int i = 0; i < nrows; ++i)
        acc = fmaf(r[(size_t)i * KCLUST + k], data[(size_t)i * DOUT + d], acc);
    cmean[k * DOUT + d] = acc;
}

__global__ void k_divide(const float* __restrict__ cmean, const float* __restrict__ cr,
                         float* __restrict__ mu, int total)
{
    const int e = blockIdx.x * blockDim.x + threadIdx.x;
    if (e < total) mu[e] = cmean[e] / cr[e >> 7];   // DOUT == 128
}

// ---------------------------------------------------------------------------
extern "C" void kernel_launch(void* const* d_in, const int* in_sizes, int n_in,
                              void* d_out, int out_size, void* d_ws, size_t ws_size,
                              hipStream_t stream)
{
    const float* x       = (const float*)d_in[0];   // [10000,1024]
    const float* adj     = (const float*)d_in[1];   // [10000,10000]
    const float* W1      = (const float*)d_in[2];   // [1024,512]
    const float* b1      = (const float*)d_in[3];   // [512]
    const float* W2      = (const float*)d_in[4];   // [512,128]
    const float* b2      = (const float*)d_in[5];   // [128]
    const float* init_mu = (const float*)d_in[6];   // [50,128]
    // d_in[7] = num_iter: fixed to 10 by the harness; loop count must be
    // host-known (no sync/copies allowed under graph capture).
    (void)in_sizes; (void)n_in; (void)out_size; (void)ws_size;

    constexpr int N = 10000, NFEAT = 1024, NHID = 512, NOUT = 128, ITERS = 10;
    constexpr int KP = 10016;                       // 10000 rounded up to 32

    float* out_mu     = (float*)d_out;              // [50,128]
    float* out_r      = out_mu + KCLUST * NOUT;     // [10000,50]
    float* out_embeds = out_r + (size_t)N * KCLUST; // [10000,128]
    float* out_dist   = out_embeds + (size_t)N * NOUT; // [10000,50]

    // workspace bump allocator (256 B aligned)
    char* ws = (char*)d_ws;
    size_t off = 0;
    auto alloc = [&](size_t bytes) -> char* {
        char* p = ws + off;
        off += (bytes + 255) & ~(size_t)255;
        return p;
    };
    __bf16* W1T  = (__bf16*)alloc((size_t)NHID * NFEAT * 2); // [512,1024]
    __bf16* W2T  = (__bf16*)alloc((size_t)NOUT * NHID * 2);  // [128,512]
    __bf16* T1T  = (__bf16*)alloc((size_t)NHID * KP * 2);    // (x@W1)^T padded
    __bf16* HW2T = (__bf16*)alloc((size_t)NOUT * KP * 2);    // (H@W2)^T padded
    float*  H    = (float*)alloc((size_t)N * NHID * 4);
    float*  data = (float*)alloc((size_t)N * NOUT * 4);
    float*  rtmp = (float*)alloc((size_t)N * KCLUST * 4);
    float*  mu_a = (float*)alloc((size_t)KCLUST * NOUT * 4);
    float*  mu_b = (float*)alloc((size_t)KCLUST * NOUT * 4);
    float*  cmean= (float*)alloc((size_t)KCLUST * NOUT * 4);
    float*  cr   = (float*)alloc(256);

    // zero the K-padding of the transposed intermediates (written only m<N)
    hipMemsetAsync(T1T,  0, (size_t)NHID * KP * 2, stream);
    hipMemsetAsync(HW2T, 0, (size_t)NOUT * KP * 2, stream);

    // stage weights transposed to bf16
    {
        long tot = (long)NHID * NFEAT;
        k_transpose_bf16<<<(tot + 255) / 256, 256, 0, stream>>>(W1, W1T, NFEAT, NHID, NFEAT);
    }
    {
        long tot = (long)NOUT * NHID;
        k_transpose_bf16<<<(tot + 255) / 256, 256, 0, stream>>>(W2, W2T, NHID, NOUT, NHID);
    }

    dim3 blk(256);
    // T1 = x @ W1 (stored transposed bf16, padded). N=512: <1,8> -> block 16x512,
    // grid.y == 1 so x is streamed from DRAM exactly once.
    {
        dim3 g((N + 15) / 16, NHID / (8 * 64));
        wmma_gemm<1, 8><<<g, blk, 0, stream>>>(x, W1T, nullptr, nullptr, T1T,
                                               N, NHID, NFEAT, NFEAT, 0, KP, 0, 1);
    }
    // H = relu(adj @ T1 + b1)  (f32 row-major). N=512: <1,8>, adj read once.
    {
        dim3 g((N + 15) / 16, NHID / (8 * 64));
        wmma_gemm<1, 8><<<g, blk, 0, stream>>>(adj, T1T, b1, H, nullptr,
                                               N, NHID, N, KP, NHID, 0, 1, 0);
    }
    // HW2 = H @ W2 (stored transposed bf16, padded). N=128: <4,2> -> block 64x128.
    {
        dim3 g((N + 63) / 64, NOUT / (2 * 64));
        wmma_gemm<4, 2><<<g, blk, 0, stream>>>(H, W2T, nullptr, nullptr, HW2T,
                                               N, NOUT, NHID, NHID, 0, KP, 0, 1);
    }
    // embeds = adj @ HW2 + b2  (f32 straight into d_out). N=128: <4,2>, adj once.
    {
        dim3 g((N + 63) / 64, NOUT / (2 * 64));
        wmma_gemm<4, 2><<<g, blk, 0, stream>>>(adj, HW2T, b2, out_embeds, nullptr,
                                               N, NOUT, N, KP, NOUT, 0, 0, 0);
    }

    // row-normalize embeddings
    k_normalize<<<N, DOUT, 0, stream>>>(out_embeds, data);

    // k-means: mu_a <- init_mu, 10 ping-pong iterations
    hipMemcpyAsync(mu_a, init_mu, (size_t)KCLUST * NOUT * 4,
                   hipMemcpyDeviceToDevice, stream);
    float* cur = mu_a;
    float* nxt = mu_b;
    const int agrid = (N + 255) / 256;
    for (int it = 0; it < ITERS; ++it) {
        k_assign<<<agrid, 256, 0, stream>>>(data, cur, rtmp, nullptr, N);
        k_cluster_r<<<KCLUST, 256, 0, stream>>>(rtmp, cr, N);
        k_cluster_mean<<<KCLUST, DOUT, 0, stream>>>(rtmp, data, cmean, N);
        k_divide<<<(KCLUST * NOUT + 255) / 256, 256, 0, stream>>>(cmean, cr, nxt, KCLUST * NOUT);
        float* t = cur; cur = nxt; nxt = t;
    }
    // one more step from mu_init -> final mu (into d_out)
    k_assign<<<agrid, 256, 0, stream>>>(data, cur, rtmp, nullptr, N);
    k_cluster_r<<<KCLUST, 256, 0, stream>>>(rtmp, cr, N);
    k_cluster_mean<<<KCLUST, DOUT, 0, stream>>>(rtmp, data, cmean, N);
    k_divide<<<(KCLUST * NOUT + 255) / 256, 256, 0, stream>>>(cmean, cr, out_mu, KCLUST * NOUT);
    // final dist + r with the new mu (into d_out)
    k_assign<<<agrid, 256, 0, stream>>>(data, out_mu, out_r, out_dist, N);
}
